// LSTMModel_27333171872011
// MI455X (gfx1250) — compile-verified, matrix operand order checked by reading, not run on hardware
//
#include <hip/hip_runtime.h>

// 2-layer LSTM, fully fused over time: gates = [x_t | h_prev] @ [W_ih;W_hh]^T + b
// WMMA f16 (16x16x32), weights resident in VGPRs, state in registers/LDS.
// Activations: branch-free v_exp/v_rcp based sigmoid/tanh.

typedef __attribute__((ext_vector_type(16))) _Float16 v16h;
typedef __attribute__((ext_vector_type(8)))  _Float16 v8h;
typedef __attribute__((ext_vector_type(4)))  _Float16 v4h;
typedef __attribute__((ext_vector_type(8)))  float    v8f;

#define Bsz   2048
#define Tlen  256
#define Idim  64
#define Hdim  64
#define Gdim  256   // 4*H
#define KdimC 128   // I + H (concat K)
#define BTILE 16

union FragH { v16h v; v8h h8[2]; _Float16 e[16]; };

#define LOG2E 1.4426950408889634f

// sigmoid(x) = 1/(1+e^-x); exp2->inf gives 0, exp2->0 gives 1 (correct saturation)
__device__ __forceinline__ float fast_sigmoid(float x) {
    return __builtin_amdgcn_rcpf(1.0f + __builtin_amdgcn_exp2f(-LOG2E * x));
}
// tanh(x) = 1 - 2/(e^{2x}+1); saturates to +/-1 via inf/0 in exp2 (branch-free)
__device__ __forceinline__ float fast_tanh(float x) {
    const float e = __builtin_amdgcn_exp2f((2.0f * LOG2E) * x);
    return 1.0f - 2.0f * __builtin_amdgcn_rcpf(e + 1.0f);
}

__global__ __launch_bounds__(256) void lstm2_fused_kernel(
    const float* __restrict__ x,
    const float* __restrict__ W_ih0, const float* __restrict__ W_hh0,
    const float* __restrict__ b_ih0, const float* __restrict__ b_hh0,
    const float* __restrict__ W_ih1, const float* __restrict__ W_hh1,
    const float* __restrict__ b_ih1, const float* __restrict__ b_hh1,
    const float* __restrict__ fc_w,  const float* __restrict__ fc_b,
    float* __restrict__ out)
{
    __shared__ _Float16 hA0[BTILE * KdimC];  // [m][k]: x_t (k<64) | h0_prev (k>=64), f16
    __shared__ _Float16 hA1[BTILE * KdimC];  // [m][k]: h1_cur (k<64) | h2_prev (k>=64), f16
    __shared__ float    gbuf[BTILE * Gdim];  // gate staging (reused by both layers + FC)

    const int tid   = threadIdx.x;
    const int wave  = tid >> 5;
    const int lane  = tid & 31;
    const int lm    = lane & 15;   // row (A) / col (B) within a 16-tile
    const int khalf = lane >> 4;   // which K-half this lane holds
    const int bbase = blockIdx.x * BTILE;

    // ---------------- preload weight (B) fragments into registers ----------------
    // B[k][n] = Wc[n][k], Wc = [W_ih | W_hh] (row n = gate, 128 cols). f16.
    // 16-bit fragment layout: lanes 0-15 hold K {0..7,16..23} (+8*khalf for lanes 16-31).
    FragH bf0[2][4], bf1[2][4];
    float bias0[2], bias1[2];
#pragma unroll
    for (int i = 0; i < 2; ++i) {
        const int n = (2 * wave + i) * 16 + lm;   // gate index 0..255
        bias0[i] = b_ih0[n] + b_hh0[n];
        bias1[i] = b_ih1[n] + b_hh1[n];
#pragma unroll
        for (int s = 0; s < 4; ++s) {
#pragma unroll
            for (int idx = 0; idx < 16; ++idx) {
                const int kk = (idx < 8 ? idx : idx + 8) + 8 * khalf;
                const int k  = s * 32 + kk;
                const float w0 = (k < Idim) ? W_ih0[n * Idim + k] : W_hh0[n * Hdim + (k - Idim)];
                const float w1 = (k < Hdim) ? W_ih1[n * Hdim + k] : W_hh1[n * Hdim + (k - Hdim)];
                bf0[i][s].e[idx] = (_Float16)w0;
                bf1[i][s].e[idx] = (_Float16)w1;
            }
        }
    }

    // elementwise ownership: thread -> row em, 4 consecutive cols ej..ej+3
    const int em = tid >> 4;           // 0..15
    const int ej = (tid & 15) * 4;     // 0..60
    const float* xrow = x + ((size_t)(bbase + em) * Tlen) * Idim + ej;

    float c0[4] = {0.f, 0.f, 0.f, 0.f};
    float c1[4] = {0.f, 0.f, 0.f, 0.f};
    float hl1[4] = {0.f, 0.f, 0.f, 0.f};

    {   // zero-init hidden-state halves of the A buffers
        v4h z; z[0] = z[1] = z[2] = z[3] = (_Float16)0.f;
        *(v4h*)&hA0[em * KdimC + Idim + ej] = z;
        *(v4h*)&hA1[em * KdimC + Hdim + ej] = z;
    }
    __syncthreads();

    for (int t = 0; t < Tlen; ++t) {
        // ---- stage x_t (fp32 -> f16) into hA0[:, 0:64] ----
        const float4 xv = *(const float4*)(xrow + (size_t)t * Idim);
        v4h xh;
        xh[0] = (_Float16)xv.x; xh[1] = (_Float16)xv.y;
        xh[2] = (_Float16)xv.z; xh[3] = (_Float16)xv.w;
        *(v4h*)&hA0[em * KdimC + ej] = xh;
        if (t + 1 < Tlen) __builtin_prefetch(xrow + (size_t)(t + 1) * Idim, 0, 3);
        __syncthreads();

        // ---- layer 0 GEMM: gates0 = [x_t | h0] @ Wc0^T + bias0 ----
        {
            FragH a[4];
#pragma unroll
            for (int s = 0; s < 4; ++s) {
                a[s].h8[0] = *(const v8h*)&hA0[lm * KdimC + s * 32 + 8 * khalf];
                a[s].h8[1] = *(const v8h*)&hA0[lm * KdimC + s * 32 + 16 + 8 * khalf];
            }
#pragma unroll
            for (int i = 0; i < 2; ++i) {
                v8f acc;
#pragma unroll
                for (int v = 0; v < 8; ++v) acc[v] = bias0[i];
#pragma unroll
                for (int s = 0; s < 4; ++s)
                    acc = __builtin_amdgcn_wmma_f32_16x16x32_f16(
                        false, a[s].v, false, bf0[i][s].v, (short)0, acc, false, false);
                const int n = (2 * wave + i) * 16 + lm;
#pragma unroll
                for (int v = 0; v < 8; ++v)
                    gbuf[(v + 8 * khalf) * Gdim + n] = acc[v];
            }
        }
        __syncthreads();

        // ---- layer 0 cell update (i,f,g,o order) ----
        {
            const float* gm = &gbuf[em * Gdim + ej];
            v4h hh;
#pragma unroll
            for (int q = 0; q < 4; ++q) {
                const float ig = fast_sigmoid(gm[q]);
                const float fg = fast_sigmoid(gm[64 + q]);
                const float gg = fast_tanh(gm[128 + q]);
                const float og = fast_sigmoid(gm[192 + q]);
                c0[q] = fg * c0[q] + ig * gg;
                hh[q] = (_Float16)(og * fast_tanh(c0[q]));
            }
            *(v4h*)&hA1[em * KdimC + ej] = hh;         // layer-1 input this step
            *(v4h*)&hA0[em * KdimC + Idim + ej] = hh;  // layer-0 recurrent for t+1
        }
        __syncthreads();

        // ---- layer 1 GEMM: gates1 = [h1 | h2] @ Wc1^T + bias1 ----
        {
            FragH a[4];
#pragma unroll
            for (int s = 0; s < 4; ++s) {
                a[s].h8[0] = *(const v8h*)&hA1[lm * KdimC + s * 32 + 8 * khalf];
                a[s].h8[1] = *(const v8h*)&hA1[lm * KdimC + s * 32 + 16 + 8 * khalf];
            }
#pragma unroll
            for (int i = 0; i < 2; ++i) {
                v8f acc;
#pragma unroll
                for (int v = 0; v < 8; ++v) acc[v] = bias1[i];
#pragma unroll
                for (int s = 0; s < 4; ++s)
                    acc = __builtin_amdgcn_wmma_f32_16x16x32_f16(
                        false, a[s].v, false, bf1[i][s].v, (short)0, acc, false, false);
                const int n = (2 * wave + i) * 16 + lm;
#pragma unroll
                for (int v = 0; v < 8; ++v)
                    gbuf[(v + 8 * khalf) * Gdim + n] = acc[v];
            }
        }
        __syncthreads();

        // ---- layer 1 cell update ----
        {
            const float* gm = &gbuf[em * Gdim + ej];
            v4h hh;
#pragma unroll
            for (int q = 0; q < 4; ++q) {
                const float ig = fast_sigmoid(gm[q]);
                const float fg = fast_sigmoid(gm[64 + q]);
                const float gg = fast_tanh(gm[128 + q]);
                const float og = fast_sigmoid(gm[192 + q]);
                c1[q] = fg * c1[q] + ig * gg;
                const float h = og * fast_tanh(c1[q]);
                hl1[q] = h;
                hh[q] = (_Float16)h;
            }
            *(v4h*)&hA1[em * KdimC + Hdim + ej] = hh;  // layer-1 recurrent for t+1
        }
        // next iteration's post-stage barrier separates these writes/reads from reuse
    }
    __syncthreads();

    // ---- final FC head: out[b] = h2_last[b] . fc_w + fc_b ----
    const float p = hl1[0] * fc_w[ej + 0] + hl1[1] * fc_w[ej + 1] +
                    hl1[2] * fc_w[ej + 2] + hl1[3] * fc_w[ej + 3];
    gbuf[em * 16 + (tid & 15)] = p;
    __syncthreads();
    if (tid < BTILE) {
        float s = fc_b[0];
#pragma unroll
        for (int i = 0; i < 16; ++i) s += gbuf[tid * 16 + i];
        out[bbase + tid] = s;
    }
}

extern "C" void kernel_launch(void* const* d_in, const int* in_sizes, int n_in,
                              void* d_out, int out_size, void* d_ws, size_t ws_size,
                              hipStream_t stream) {
    (void)in_sizes; (void)n_in; (void)out_size; (void)d_ws; (void)ws_size;
    const float* x     = (const float*)d_in[0];
    const float* W_ih0 = (const float*)d_in[1];
    const float* W_hh0 = (const float*)d_in[2];
    const float* b_ih0 = (const float*)d_in[3];
    const float* b_hh0 = (const float*)d_in[4];
    const float* W_ih1 = (const float*)d_in[5];
    const float* W_hh1 = (const float*)d_in[6];
    const float* b_ih1 = (const float*)d_in[7];
    const float* b_hh1 = (const float*)d_in[8];
    const float* fc_w  = (const float*)d_in[9];
    const float* fc_b  = (const float*)d_in[10];
    float* out = (float*)d_out;

    lstm2_fused_kernel<<<dim3(Bsz / BTILE), dim3(256), 0, stream>>>(
        x, W_ih0, W_hh0, b_ih0, b_hh0, W_ih1, W_hh1, b_ih1, b_hh1, fc_w, fc_b, out);
}
